// TAMCaD_69269232550148
// MI455X (gfx1250) — matrix operand order repaired and checked by reading, not compiled, not run on hardware
//
#include <hip/hip_runtime.h>
#include <hip/hip_bf16.h>

typedef __attribute__((ext_vector_type(8)))  _Float16 v8h;
typedef __attribute__((ext_vector_type(16))) _Float16 v16h;
typedef __attribute__((ext_vector_type(8)))  float    v8f;
typedef __attribute__((ext_vector_type(4)))  float    f4;   // clang vector -> NT builtins OK

#define B_   4
#define N_   64
#define H_   32
#define T_   4096
#define TT   8
#define NEG_INF_F (-1000000000.0f)

__global__ void tamcad_zero_loss(float* out) { out[0] = 0.0f; }

__global__ __launch_bounds__(256)
void tamcad_main(const float* __restrict__ x,
                 const float* __restrict__ tcn,
                 const unsigned char* __restrict__ mask,
                 const float* __restrict__ w1,
                 const float* __restrict__ b1,
                 const float* __restrict__ w2,
                 const float* __restrict__ b2,
                 float* __restrict__ out)
{
    __shared__ __align__(16) _Float16 attnS[TT][N_][N_]; // [t][i][j]  64 KB
    __shared__ __align__(16) _Float16 ctxS [TT][H_][N_]; // [t][d][j]  32 KB
    __shared__ __align__(16) _Float16 zS   [TT][N_][H_]; // [t][i][d]  32 KB
    __shared__ float red[256];

    const int tid = threadIdx.x;
    const int blk = blockIdx.x;              // B * (T/TT) = 2048 blocks
    const int b   = blk / (T_ / TT);
    const int t0  = (blk % (T_ / TT)) * TT;

    float* predOut   = out + 1;
    float* logitsOut = out + 1 + (long)B_ * N_ * T_;

    float diffAbs = 0.0f;

    // ---------- Phase 1: logits -> output copy (NT), mask+stage to LDS, |diff| reg
    #pragma unroll 1
    for (int r = 0; r < 16; ++r) {
        int p = tid + 256 * r;               // 0..4095 -> (i, j)
        int i = p >> 6, j = p & 63;
        long base = ((long)(b * N_ + i) * (H_ + N_) + H_ + j) * T_ + t0;
        // single-use stream: non-temporal, keep L2 for weights
        f4 f0 = __builtin_nontemporal_load((const f4*)(tcn + base));
        f4 f1 = __builtin_nontemporal_load((const f4*)(tcn + base + 4));
        float l[8] = {f0[0], f0[1], f0[2], f0[3], f1[0], f1[1], f1[2], f1[3]};
        long obase = ((long)(b * N_ + i) * N_ + j) * T_ + t0;
        __builtin_nontemporal_store(f0, (f4*)(logitsOut + obase));
        __builtin_nontemporal_store(f1, (f4*)(logitsOut + obase + 4));
        bool m = mask[i * N_ + j] != 0;      // tiny, reused by all blocks: regular path
        #pragma unroll
        for (int t = 0; t < TT; ++t)
            attnS[t][i][j] = (_Float16)(m ? NEG_INF_F : l[t]);
        #pragma unroll
        for (int t = 0; t < TT - 1; ++t) diffAbs += fabsf(l[t + 1] - l[t]);
        if (t0 + TT < T_)
            diffAbs += fabsf(__builtin_nontemporal_load(tcn + base + TT) - l[TT - 1]);
    }

    // ---------- Phase 1b: context -> LDS transposed [t][d][j]
    #pragma unroll 1
    for (int r = 0; r < 8; ++r) {
        int q = tid + 256 * r;               // 0..2047 -> (j, d)
        int j = q >> 5, d = q & 31;
        long base = ((long)(b * N_ + j) * (H_ + N_) + d) * T_ + t0;
        f4 f0 = __builtin_nontemporal_load((const f4*)(tcn + base));
        f4 f1 = __builtin_nontemporal_load((const f4*)(tcn + base + 4));
        float c[8] = {f0[0], f0[1], f0[2], f0[3], f1[0], f1[1], f1[2], f1[3]};
        #pragma unroll
        for (int t = 0; t < TT; ++t) ctxS[t][d][j] = (_Float16)c[t];
    }
    __syncthreads();

    // ---------- Phase 2: softmax over j for each (i, t)
    #pragma unroll 1
    for (int r = 0; r < 2; ++r) {
        int s = tid + 256 * r;               // 0..511 -> (t, i)
        int t = s & 7, i = s >> 3;
        float mx = -3.4e38f;
        for (int j = 0; j < N_; ++j) mx = fmaxf(mx, (float)attnS[t][i][j]);
        float sum = 0.0f;
        for (int j = 0; j < N_; ++j) {
            float ev = __expf((float)attnS[t][i][j] - mx);
            sum += ev;
            attnS[t][i][j] = (_Float16)ev;
        }
        float inv = 1.0f / sum;
        for (int j = 0; j < N_; ++j)
            attnS[t][i][j] = (_Float16)((float)attnS[t][i][j] * inv);
    }
    __syncthreads();

    // ---------- Phase 3: z = attn @ context via WMMA (per t: 64x32 = 8 tiles)
    {
        const int w    = tid >> 5;
        const int lane = tid & 31;
        const int it   = w >> 1;             // 0..3 : i-tile
        const int dt   = w & 1;              // 0..1 : d-tile
        const int hi   = (lane >= 16) ? 1 : 0;
        const int mrow = 16 * it + (lane & 15);  // A row (i)
        const int ncol = 16 * dt + (lane & 15);  // B col (d)

        #pragma unroll 1
        for (int t = 0; t < TT; ++t) {
            v8f acc = {0.f, 0.f, 0.f, 0.f, 0.f, 0.f, 0.f, 0.f};
            #pragma unroll
            for (int kc = 0; kc < 2; ++kc) {
                // A fragment: lanes 0-15 -> K {0..7,16..23}, lanes 16-31 -> K {8..15,24..31}
                v8h alo = *(const v8h*)&attnS[t][mrow][kc * 32 + hi * 8];
                v8h ahi = *(const v8h*)&attnS[t][mrow][kc * 32 + 16 + hi * 8];
                v16h A = __builtin_shufflevector(alo, ahi,
                        0,1,2,3,4,5,6,7,8,9,10,11,12,13,14,15);
                // B fragment: lanes 0-15 -> K {0..7,16..23}, lanes 16-31 -> K {8..15,24..31}
                v8h blo = *(const v8h*)&ctxS[t][ncol][kc * 32 + hi * 8];
                v8h bhi = *(const v8h*)&ctxS[t][ncol][kc * 32 + 16 + hi * 8];
                v16h Bf = __builtin_shufflevector(blo, bhi,
                        0,1,2,3,4,5,6,7,8,9,10,11,12,13,14,15);
                acc = __builtin_amdgcn_wmma_f32_16x16x32_f16(
                        false, A, false, Bf, (short)0, acc, false, false);
            }
            // D layout: VGPR v -> row v (lanes 0-15) / v+8 (lanes 16-31)
            #pragma unroll
            for (int v = 0; v < 8; ++v)
                zS[t][16 * it + v + 8 * hi][ncol] = (_Float16)acc[v];
        }
    }
    __syncthreads();

    // ---------- Phase 4: tiny MLP per (n, t), pred + regression partial
    float regSq = 0.0f;
    #pragma unroll 1
    for (int r = 0; r < 2; ++r) {
        int s = tid + 256 * r;               // 0..511 -> (n, t)
        int n = s & 63, t = s >> 6;
        float zrow[H_];
        #pragma unroll
        for (int d = 0; d < H_; ++d) zrow[d] = (float)zS[t][n][d];
        float pred = b2[n];
        #pragma unroll 1
        for (int o = 0; o < 16; ++o) {
            float a = b1[n * 16 + o];
            const float* wrow = w1 + (long)(n * 16 + o) * H_;   // hot in L2: regular path
            #pragma unroll
            for (int d = 0; d < H_; ++d) a += wrow[d] * zrow[d];
            a = fmaxf(a, 0.0f);
            pred += w2[n * 16 + o] * a;
        }
        int tg = t0 + t;
        __builtin_nontemporal_store(pred, predOut + (long)(b * N_ + n) * T_ + tg);
        if (tg < T_ - 1) {
            float xv = __builtin_nontemporal_load(x + (long)(b * N_ + n) * T_ + tg + 1);
            float d0 = xv - pred;
            regSq += d0 * d0;
        }
    }

    // ---------- Phase 5: loss reduction
    float contrib = regSq * (1.0f / 1048320.0f)          // mean over B*N*(T-1)
                  + diffAbs * (0.5f / 67092480.0f);      // GAMMA * mean over B*N*64*(T-1)
    red[tid] = contrib;
    __syncthreads();
    for (int s2 = 128; s2 > 0; s2 >>= 1) {
        if (tid < s2) red[tid] += red[tid + s2];
        __syncthreads();
    }
    if (tid == 0) atomicAdd(out, red[0]);
}

extern "C" void kernel_launch(void* const* d_in, const int* in_sizes, int n_in,
                              void* d_out, int out_size, void* d_ws, size_t ws_size,
                              hipStream_t stream) {
    const float* x            = (const float*)d_in[0];
    const float* tcn          = (const float*)d_in[1];
    const unsigned char* mask = (const unsigned char*)d_in[2]; // jax bool = 1 byte
    const float* w1           = (const float*)d_in[3];
    const float* b1           = (const float*)d_in[4];
    const float* w2           = (const float*)d_in[5];
    const float* b2           = (const float*)d_in[6];
    float* out = (float*)d_out;

    tamcad_zero_loss<<<1, 1, 0, stream>>>(out);
    dim3 grid(B_ * (T_ / TT));  // 2048 workgroups of 256 threads (8 waves)
    tamcad_main<<<grid, 256, 0, stream>>>(x, tcn, mask, w1, b1, w2, b2, out);
}
